// BiGNN_82454782149147
// MI455X (gfx1250) — matrix operand order repaired and checked by reading, not compiled, not run on hardware
//
#include <hip/hip_runtime.h>
#include <hip/hip_bf16.h>

typedef __attribute__((ext_vector_type(16))) _Float16 v16h;
typedef __attribute__((ext_vector_type(8)))  float    v8f;

#define ROWS 4096   // B*N = 8*512

// ---------------------------------------------------------------------------
// Batched WMMA GEMM:  C[M,N] = act( A[M,K] * B[K,N] + bias[N] )
// A: f16 row-major [M,K].  Bt: f16 COLUMN-major image of B, i.e. Bt[n*K+k].
// C: f32 row-major. One wave computes a 64x16 strip (4 WMMA tiles), reusing
// one B fragment per K-step. M % 64 == 0, N % 16 == 0, K % 32 == 0.
// ---------------------------------------------------------------------------
template<int ACT, int BIAS>
__global__ void k_gemm(const _Float16* __restrict__ A,
                       const _Float16* __restrict__ Bt,
                       const float* __restrict__ bias,
                       float* __restrict__ C,
                       int M, int N, int K,
                       long long sA, long long sB, long long sC)
{
    int wave = threadIdx.x >> 5;
    int lane = threadIdx.x & 31;
    int stripsN = N >> 4;
    int totalStrips = (M >> 6) * stripsN;
    int strip = blockIdx.x * (blockDim.x >> 5) + wave;
    if (strip >= totalStrips) return;        // whole-wave exit: EXEC all-1 for WMMA
    int sm = strip / stripsN;
    int tn = strip % stripsN;

    long long batch = blockIdx.y;
    A += batch * sA;  Bt += batch * sB;  C += batch * sC;

    int  laneLo = lane & 15;
    bool hi     = lane >= 16;
    int  n  = tn * 16 + laneLo;              // B/C column for this lane
    int  m0 = sm * 64 + laneLo;              // A row (first sub-tile) for this lane

    // B fragment: lanes0-15 K=k0..k0+15, lanes16-31 K=k0+16..k0+31 (contiguous)
    const _Float16* bptr = Bt + (size_t)n * K + (hi ? 16 : 0);

    v8f acc[4] = {};
    for (int k0 = 0; k0 < K; k0 += 32) {
        v16h vb;
#pragma unroll
        for (int e = 0; e < 16; ++e) vb[e] = bptr[k0 + e];

        __builtin_prefetch(bptr + k0 + 32, 0, 3);
        __builtin_prefetch(A + (size_t)m0 * K + k0 + 32, 0, 3);

#pragma unroll
        for (int sub = 0; sub < 4; ++sub) {
            // A 16x32 f16 layout: lanes0-15 hold K=0..15, lanes16-31 K=16..31
            // (contiguous, matching B).
            const _Float16* aptr = A + (size_t)(m0 + sub * 16) * K + k0 + (hi ? 16 : 0);
            v16h va;
#pragma unroll
            for (int v = 0; v < 8; ++v) {
                int kA = 2 * v;
                va[2 * v]     = aptr[kA];
                va[2 * v + 1] = aptr[kA + 1];
            }
            acc[sub] = __builtin_amdgcn_wmma_f32_16x16x32_f16(false, va, false, vb,
                                                              (short)0, acc[sub],
                                                              false, false);
        }
    }

    float bv = 0.f;
    if (BIAS) bv = bias[n];
#pragma unroll
    for (int sub = 0; sub < 4; ++sub) {
#pragma unroll
        for (int r = 0; r < 8; ++r) {
            int row = sm * 64 + sub * 16 + r + (hi ? 8 : 0);
            float val = acc[sub][r];
            if (BIAS) val += bv;
            if (ACT == 1) val = (val > 0.f) ? val : 0.01f * val;
            C[(size_t)row * N + n] = val;
        }
    }
}

// ---------------------------------------------------------------------------
__global__ void k_cvt(const float* __restrict__ src, _Float16* __restrict__ dst, int n)
{
    int i = blockIdx.x * blockDim.x + threadIdx.x;
    if (i < n) dst[i] = (_Float16)src[i];
}

// src: f32 [B,K,N] row-major weights -> dst: f16 [B,N,K] (column-major images)
__global__ void k_cvt_t(const float* __restrict__ src, _Float16* __restrict__ dst,
                        int K, int N, int total)
{
    int i = blockIdx.x * blockDim.x + threadIdx.x;
    if (i >= total) return;
    int kn  = K * N;
    int mat = i / kn;
    int rem = i - mat * kn;
    int k = rem / N, n = rem - k * N;
    dst[(size_t)mat * kn + (size_t)n * K + k] = (_Float16)src[i];
}

// Wh f32 [h, b*512+j, o] -> WhT f16 [(h*8+b), o, j]   (col-major B for hp GEMM)
__global__ void k_wht(const float* __restrict__ Wh, _Float16* __restrict__ WhT, int total)
{
    int i = blockIdx.x * blockDim.x + threadIdx.x;
    if (i >= total) return;
    int h   = i >> 19;                    // / (4096*128)
    int rem = i & ((1 << 19) - 1);
    int r   = rem >> 7;                   // b*512 + j
    int o   = rem & 127;
    int b   = r >> 9;
    int j   = r & 511;
    WhT[(((size_t)(h * 8 + b)) * 128 + o) * 512 + j] = (_Float16)Wh[i];
}

// out = layernorm(a + b) over rows of 128; also emit f16 copy for next GEMM
__global__ void k_add_ln(const float* __restrict__ a, const float* __restrict__ b,
                         float* __restrict__ out, _Float16* __restrict__ out16)
{
    __shared__ float red[128];
    int row = blockIdx.x;
    int t = threadIdx.x;
    size_t idx = (size_t)row * 128 + t;
    float x = a[idx] + b[idx];
    red[t] = x; __syncthreads();
    for (int s = 64; s > 0; s >>= 1) { if (t < s) red[t] += red[t + s]; __syncthreads(); }
    float mean = red[0] * (1.f / 128.f);
    __syncthreads();
    float c = x - mean;
    red[t] = c * c; __syncthreads();
    for (int s = 64; s > 0; s >>= 1) { if (t < s) red[t] += red[t + s]; __syncthreads(); }
    float var = red[0] * (1.f / 128.f);
    float y = c * rsqrtf(var + 1e-5f);
    out[idx]   = y;
    out16[idx] = (_Float16)y;
}

// s1[h,r] = Wh[h,r,:]·a[h,:128], s2 = Wh·a[h,128:256]
__global__ void k_s1s2(const float* __restrict__ Wh, const float* __restrict__ a,
                       float* __restrict__ s1, float* __restrict__ s2, int total)
{
    int t = blockIdx.x * blockDim.x + threadIdx.x;
    if (t >= total) return;
    int h = t >> 12;                       // t / 4096
    const float* a1 = a + (size_t)h * 256;
    const float* a2 = a1 + 128;
    const float* w  = Wh + (size_t)t * 128;
    float x1 = 0.f, x2 = 0.f;
    for (int o = 0; o < 128; ++o) { float v = w[o]; x1 += v * a1[o]; x2 += v * a2[o]; }
    s1[t] = x1; s2[t] = x2;
}

// rank-1 masked-softmax attention row -> f16. One block per (h,b,i) row, 256 thr.
__global__ void k_softmax(const float* __restrict__ s1, const float* __restrict__ s2,
                          const int* __restrict__ adj, _Float16* __restrict__ att,
                          int transposed)
{
    __shared__ float red[256];
    int row = blockIdx.x;                  // h*4096 + b*512 + i
    int h = row >> 12;
    int b = (row >> 9) & 7;
    int i = row & 511;
    int t = threadIdx.x;
    float s1v = s1[row];
    const int* adjb = adj + (size_t)b * 512 * 512;
    const float* s2b = s2 + (size_t)h * 4096 + (size_t)b * 512;

    int j0 = t, j1 = t + 256;
    int m0 = transposed ? adjb[(size_t)j0 * 512 + i] : adjb[(size_t)i * 512 + j0];
    int m1 = transposed ? adjb[(size_t)j1 * 512 + i] : adjb[(size_t)i * 512 + j1];
    float v0 = s1v * s2b[j0]; v0 = (v0 > 0.f) ? v0 : 0.01f * v0;
    float v1 = s1v * s2b[j1]; v1 = (v1 > 0.f) ? v1 : 0.01f * v1;
    float e0 = (m0 > 0) ? v0 : -9e15f;
    float e1 = (m1 > 0) ? v1 : -9e15f;

    red[t] = fmaxf(e0, e1); __syncthreads();
    for (int s = 128; s > 0; s >>= 1) { if (t < s) red[t] = fmaxf(red[t], red[t + s]); __syncthreads(); }
    float M = red[0];
    __syncthreads();
    float p0 = expf(e0 - M);
    float p1 = expf(e1 - M);
    red[t] = p0 + p1; __syncthreads();
    for (int s = 128; s > 0; s >>= 1) { if (t < s) red[t] += red[t + s]; __syncthreads(); }
    float inv = 1.f / red[0];
    size_t base = (size_t)row * 512;
    att[base + j0] = (_Float16)((m0 > 0) ? p0 * inv : 0.f);
    att[base + j1] = (_Float16)((m1 > 0) ? p1 * inv : 0.f);
}

// hp[h, b*512+n, o] --elu--> cat[b*512+n, h*128+o] (f16, A operand: row-major)
__global__ void k_cat_elu(const float* __restrict__ hp, _Float16* __restrict__ cat)
{
    int i = blockIdx.x * blockDim.x + threadIdx.x;
    if (i >= 4 * ROWS * 128) return;
    int h   = i >> 19;                    // / (4096*128)
    int rem = i & ((1 << 19) - 1);
    int r   = rem >> 7;
    int o   = rem & 127;
    float v = hp[i];
    float e = (v > 0.f) ? v : expm1f(v);
    cat[(size_t)r * 512 + h * 128 + o] = (_Float16)e;
}

// z = mean_n concat(h1, ht); out[b] = z·projW + projb
__global__ void k_finalize(const float* __restrict__ h1, const float* __restrict__ ht,
                           const float* __restrict__ pW, const float* __restrict__ pb,
                           float* __restrict__ out)
{
    __shared__ float red[256];
    int b = blockIdx.x, t = threadIdx.x;
    const float* src = (t < 128) ? h1 : ht;
    int ch = t & 127;
    float s = 0.f;
    for (int n = 0; n < 512; ++n) s += src[((size_t)b * 512 + n) * 128 + ch];
    red[t] = (s * (1.f / 512.f)) * pW[t];
    __syncthreads();
    for (int st = 128; st > 0; st >>= 1) { if (t < st) red[t] += red[t + st]; __syncthreads(); }
    if (t == 0) out[b] = red[0] + pb[0];
}

// ---------------------------------------------------------------------------
extern "C" void kernel_launch(void* const* d_in, const int* in_sizes, int n_in,
                              void* d_out, int out_size, void* d_ws, size_t ws_size,
                              hipStream_t stream)
{
    (void)in_sizes; (void)n_in; (void)out_size; (void)ws_size;
    const float* x    = (const float*)d_in[0];
    const int*   adj  = (const int*)  d_in[1];
    const float* f1w1 = (const float*)d_in[3];  const float* f1b1 = (const float*)d_in[4];
    const float* f1w2 = (const float*)d_in[5];  const float* f1b2 = (const float*)d_in[6];
    const float* f2w1 = (const float*)d_in[7];  const float* f2b1 = (const float*)d_in[8];
    const float* f2w2 = (const float*)d_in[9];  const float* f2b2 = (const float*)d_in[10];
    const float* f3w1 = (const float*)d_in[11]; const float* f3b1 = (const float*)d_in[12];
    const float* f3w2 = (const float*)d_in[13]; const float* f3b2 = (const float*)d_in[14];
    const float* attW = (const float*)d_in[15]; const float* attA = (const float*)d_in[16];
    const float* reshW= (const float*)d_in[17]; const float* reshB= (const float*)d_in[18];
    const float* outW = (const float*)d_in[19]; const float* outA = (const float*)d_in[20];
    const float* projW= (const float*)d_in[21]; const float* projB= (const float*)d_in[22];

    // ---- workspace layout (256B aligned slices) ----
    char* ws = (char*)d_ws;
    size_t off = 0;
    auto take = [&](size_t bytes) -> void* {
        void* p = ws + off;
        off += (bytes + 255) & ~(size_t)255;
        return p;
    };
    // f16 weight arena offsets (in halves); all stored TRANSPOSED ([N,K])
    const size_t O_F1W1 = 0;
    const size_t O_F1W2 = O_F1W1 + 32 * 128;
    const size_t O_F2W1 = O_F1W2 + 128 * 128;
    const size_t O_F2W2 = O_F2W1 + 128 * 256;
    const size_t O_F3W1 = O_F2W2 + 256 * 128;
    const size_t O_F3W2 = O_F3W1 + 128 * 256;
    const size_t O_ATTW = O_F3W2 + 256 * 128;
    const size_t O_RESHW= O_ATTW + (size_t)6 * 4 * 128 * 128;
    const size_t O_OUTW = O_RESHW + (size_t)6 * 512 * 128;
    const size_t W16_TOT= O_OUTW + (size_t)2 * 128 * 128;

    _Float16* w16   = (_Float16*)take(W16_TOT * 2);
    _Float16* act16 = (_Float16*)take((size_t)ROWS * 256 * 2);
    float*    bufA  = (float*)   take((size_t)ROWS * 256 * 4);
    float*    resb  = (float*)   take((size_t)ROWS * 128 * 4);
    float*    cur   = (float*)   take((size_t)ROWS * 128 * 4);
    float*    tmp   = (float*)   take((size_t)ROWS * 128 * 4);
    float*    hb1   = (float*)   take((size_t)ROWS * 128 * 4);
    float*    Whf   = (float*)   take((size_t)4 * ROWS * 128 * 4);  // also hp f32
    _Float16* Wh16  = (_Float16*)take((size_t)4 * ROWS * 128 * 2);  // WhT (col-major)
    float*    s1    = (float*)   take((size_t)4 * ROWS * 4);
    float*    s2    = (float*)   take((size_t)4 * ROWS * 4);
    _Float16* att16 = (_Float16*)take((size_t)4 * 8 * 512 * 512 * 2);
    _Float16* cat16 = (_Float16*)take((size_t)ROWS * 512 * 2);

    auto cvt = [&](const float* s, _Float16* d, int n) {
        k_cvt<<<(n + 255) / 256, 256, 0, stream>>>(s, d, n);
    };
    auto cvtT = [&](const float* s, _Float16* d, int K, int N, int nmat) {
        int total = K * N * nmat;
        k_cvt_t<<<(total + 255) / 256, 256, 0, stream>>>(s, d, K, N, total);
    };
    auto gemm = [&](const _Float16* A, const _Float16* Bt, const float* bias, float* C,
                    int M, int N, int K, long long sA, long long sB, long long sC,
                    int batch, int act) {
        int strips = (M / 64) * (N / 16);
        dim3 g((strips + 3) / 4, batch);
        if (bias) {
            if (act) k_gemm<1, 1><<<g, 128, 0, stream>>>(A, Bt, bias, C, M, N, K, sA, sB, sC);
            else     k_gemm<0, 1><<<g, 128, 0, stream>>>(A, Bt, bias, C, M, N, K, sA, sB, sC);
        } else {
            if (act) k_gemm<1, 0><<<g, 128, 0, stream>>>(A, Bt, bias, C, M, N, K, sA, sB, sC);
            else     k_gemm<0, 0><<<g, 128, 0, stream>>>(A, Bt, bias, C, M, N, K, sA, sB, sC);
        }
    };

    // ---- stage all weights to f16, transposed to [N,K] ----
    cvtT(f1w1, w16 + O_F1W1, 32, 128, 1);
    cvtT(f1w2, w16 + O_F1W2, 128, 128, 1);
    cvtT(f2w1, w16 + O_F2W1, 128, 256, 1);
    cvtT(f2w2, w16 + O_F2W2, 256, 128, 1);
    cvtT(f3w1, w16 + O_F3W1, 128, 256, 1);
    cvtT(f3w2, w16 + O_F3W2, 256, 128, 1);
    cvtT(attW, w16 + O_ATTW, 128, 128, 24);
    cvtT(reshW, w16 + O_RESHW, 512, 128, 6);
    cvtT(outW, w16 + O_OUTW, 128, 128, 2);

    // ---- FEL (MLP front-end) ----
    cvt(x, act16, ROWS * 32);
    gemm(act16, w16 + O_F1W1, f1b1, bufA, ROWS, 128, 32, 0, 0, 0, 1, 1);
    cvt(bufA, act16, ROWS * 128);
    gemm(act16, w16 + O_F1W2, f1b2, resb, ROWS, 128, 128, 0, 0, 0, 1, 1);
    cvt(resb, act16, ROWS * 128);
    gemm(act16, w16 + O_F2W1, f2b1, bufA, ROWS, 256, 128, 0, 0, 0, 1, 1);
    cvt(bufA, act16, ROWS * 256);
    gemm(act16, w16 + O_F2W2, f2b2, tmp, ROWS, 128, 256, 0, 0, 0, 1, 1);
    k_add_ln<<<ROWS, 128, 0, stream>>>(tmp, resb, cur, act16);
    gemm(act16, w16 + O_F3W1, f3b1, bufA, ROWS, 256, 128, 0, 0, 0, 1, 1);
    cvt(bufA, act16, ROWS * 256);
    gemm(act16, w16 + O_F3W2, f3b2, tmp, ROWS, 128, 256, 0, 0, 0, 1, 0);
    k_add_ln<<<ROWS, 128, 0, stream>>>(tmp, cur, cur, act16);

    // ---- multi-head attention block ----
    auto attBlock = [&](int i, int transposed) {
        // Wh[h] = x @ W[i,h]   (batch over 4 heads, A shared)
        gemm(act16, w16 + O_ATTW + (size_t)i * 4 * 128 * 128, nullptr, Whf,
             ROWS, 128, 128, 0, 128 * 128, (long long)ROWS * 128, 4, 0);
        k_wht<<<(4 * ROWS * 128 + 255) / 256, 256, 0, stream>>>(Whf, Wh16, 4 * ROWS * 128);
        k_s1s2<<<(4 * ROWS + 255) / 256, 256, 0, stream>>>(
            Whf, attA + (size_t)i * 4 * 256, s1, s2, 4 * ROWS);
        k_softmax<<<4 * ROWS, 256, 0, stream>>>(s1, s2, adj, att16, transposed);
        // hp = att @ Wh   (batch over h*b = 32); hp reuses the dead Whf f32 buffer
        gemm(att16, Wh16, nullptr, Whf, 512, 128, 512,
             (long long)512 * 512, (long long)128 * 512, (long long)512 * 128, 32, 0);
        k_cat_elu<<<(4 * ROWS * 128 + 255) / 256, 256, 0, stream>>>(Whf, cat16);
        gemm(cat16, w16 + O_RESHW + (size_t)i * 512 * 128, reshB + (size_t)i * 128, tmp,
             ROWS, 128, 512, 0, 0, 0, 1, 0);
        k_add_ln<<<ROWS, 128, 0, stream>>>(tmp, cur, cur, act16);
    };

    // ---- single-head output GAT (concat=False, outer lrelu fused into hp GEMM) ----
    auto finalGat = [&](int which, int transposed, bool saveBranch1) {
        gemm(act16, w16 + O_OUTW + (size_t)which * 128 * 128, nullptr, Whf,
             ROWS, 128, 128, 0, 0, 0, 1, 0);
        k_wht<<<(ROWS * 128 + 255) / 256, 256, 0, stream>>>(Whf, Wh16, ROWS * 128);
        k_s1s2<<<(ROWS + 255) / 256, 256, 0, stream>>>(
            Whf, outA + (size_t)which * 256, s1, s2, ROWS);
        k_softmax<<<ROWS, 256, 0, stream>>>(s1, s2, adj, att16, transposed);
        gemm(att16, Wh16, nullptr, tmp, 512, 128, 512,
             (long long)512 * 512, (long long)128 * 512, (long long)512 * 128, 8, 1);
        if (saveBranch1) {
            hipMemcpyAsync(hb1, tmp, (size_t)ROWS * 128 * 4, hipMemcpyDeviceToDevice, stream);
            hipMemcpyAsync(cur, tmp, (size_t)ROWS * 128 * 4, hipMemcpyDeviceToDevice, stream);
            cvt(tmp, act16, ROWS * 128);
        }
    };

    // forward branch (mask = adj)
    attBlock(0, 0); attBlock(1, 0); attBlock(2, 0);
    finalGat(0, 0, true);
    // reverse branch (mask = adj^T), starting from branch-1 output
    attBlock(3, 1); attBlock(4, 1); attBlock(5, 1);
    finalGat(1, 1, false);                 // ht stays in tmp (f32)

    k_finalize<<<8, 256, 0, stream>>>(hb1, tmp, projW, projB, (float*)d_out);
}
